// GraphEGATDGL_55018531062486
// MI455X (gfx1250) — compile-verified, matrix operand order checked by reading, not run on hardware
//
#include <hip/hip_runtime.h>
#include <math.h>

#define NN 50000
#define NE 1600000

typedef _Float16 h16 __attribute__((ext_vector_type(16)));
typedef float f8 __attribute__((ext_vector_type(8)));

// ---- WMMA fragment helpers (gfx1250 wave32 layouts, CDNA5 ISA) ----
// A (f16 16x32): lane l -> row l%16; half i -> k = kb + i + (i>=8?8:0),
// kb = (l<16?0:8) + 32*kstep.  Two contiguous 8-float chunks at kb and kb+16.
__device__ __forceinline__ h16 loadAfrag(const float* row, int kb) {
  h16 a;
#pragma unroll
  for (int j = 0; j < 8; ++j) a[j] = (_Float16)row[kb + j];
#pragma unroll
  for (int j = 0; j < 8; ++j) a[8 + j] = (_Float16)row[kb + 16 + j];
  return a;
}

// B fragments are pre-packed to register layout: frag -> 32 lanes x 16 halves,
// so each lane issues one contiguous 32-byte load.
__device__ __forceinline__ h16 loadPackedB(const _Float16* __restrict__ P, int frag, int lane) {
  return *(const h16*)(P + ((size_t)frag * 32 + lane) * 16);
}

__device__ __forceinline__ f8 wmma_f16(h16 a, h16 b, f8 c) {
  return __builtin_amdgcn_wmma_f32_16x16x32_f16(false, a, false, b, (short)0, c, false, false);
}

__device__ __forceinline__ float reduce16(float v) {
  v += __shfl_xor(v, 1, 32);
  v += __shfl_xor(v, 2, 32);
  v += __shfl_xor(v, 4, 32);
  v += __shfl_xor(v, 8, 32);
  return v;
}

// ---- Pack a row-major KxN f32 weight matrix into f16 WMMA-B fragments ----
// frag f = colTile*kfrags + kstep; lane l -> col n = colTile*16 + l%16;
// half i -> k = kstep*32 + (l&16?8:0) + (i<8?i:i+8). Padded cols -> 0.
__global__ void pack_b_kernel(const float* __restrict__ Wm, _Float16* __restrict__ P,
                              int K, int N) {
  int kfrags = K >> 5;
  int ntc = (N + 15) >> 4;
  long total = (long)ntc * kfrags * 512;
  long p = (long)blockIdx.x * blockDim.x + threadIdx.x;
  if (p >= total) return;
  int i = (int)(p & 15);
  int lane = (int)((p >> 4) & 31);
  int f = (int)(p >> 9);
  int ks = f % kfrags, ct = f / kfrags;
  int n = ct * 16 + (lane & 15);
  int k = ks * 32 + ((lane & 16) ? 8 : 0) + (i < 8 ? i : i + 8);
  float v = (n < N) ? Wm[(size_t)k * N + n] : 0.0f;
  P[p] = (_Float16)v;
}

// ---- Generic WMMA GEMM: C[MxN] = act(A[MxK] @ Bpacked + bias) ----
// M % 16 == 0, K % 32 == 0; N arbitrary (store-guarded). One wave per tile.
__global__ void gemm_wmma_kernel(const float* __restrict__ A, const _Float16* __restrict__ Bp,
                                 const float* __restrict__ bias, float* __restrict__ C,
                                 int M, int N, int K, int relu) {
  int lane = threadIdx.x & 31;
  long wid = (long)(blockIdx.x * blockDim.x + threadIdx.x) >> 5;
  int ntc = (N + 15) >> 4;
  long tiles = (long)(M >> 4) * ntc;
  if (wid >= tiles) return;
  int rt = (int)(wid / ntc), ct = (int)(wid % ntc);
  int l16 = lane & 15;
  int hs = (lane & 16) ? 8 : 0;
  int kfrags = K >> 5;
  const float* aRow = A + (size_t)((unsigned)(rt * 16 + l16) * (unsigned)K);
  int n = ct * 16 + l16;
  f8 acc;
#pragma unroll
  for (int i = 0; i < 8; ++i) acc[i] = 0.0f;
  for (int ksi = 0; ksi < kfrags; ++ksi) {
    h16 af = loadAfrag(aRow, ksi * 32 + hs);
    h16 bf = loadPackedB(Bp, ct * kfrags + ksi, lane);
    acc = wmma_f16(af, bf, acc);
  }
  if (n < N) {
    float bv = bias ? bias[n] : 0.0f;
    int mb = rt * 16 + hs;
#pragma unroll
    for (int r = 0; r < 8; ++r) {
      float v = acc[r] + bv;
      if (relu) v = fmaxf(v, 0.0f);
      C[(size_t)((unsigned)(mb + r) * (unsigned)N + (unsigned)n)] = v;
    }
  }
}

// ---- Fused edge kernel: f = leaky(ef@Wfij + xWni[src] + xWnj[dst] + be) ----
// Writes he (optionally relu'd) in place, emits attention logits [E,4].
// One wave per 16 edges; col-tile t == head h (OE == 16).
__global__ void egat_edge_kernel(const float* __restrict__ he_in, int Ke,
                                 const _Float16* __restrict__ pWf,
                                 const float* __restrict__ aN, const float* __restrict__ bN,
                                 const float* __restrict__ be, const float* __restrict__ attn,
                                 const int* __restrict__ src, const int* __restrict__ dst,
                                 float* __restrict__ he_out, float* __restrict__ elog, int relu) {
  int lane = threadIdx.x & 31;
  long wid = (long)(blockIdx.x * blockDim.x + threadIdx.x) >> 5;
  if (wid >= NE / 16) return;
  unsigned e0 = (unsigned)wid * 16u;
  int l16 = lane & 15;
  int hs = (lane & 16) ? 8 : 0;
  int kfrags = Ke >> 5;
  const float* aRow = he_in + (size_t)((e0 + (unsigned)l16) * (unsigned)Ke);
  f8 acc[4];
#pragma unroll
  for (int t = 0; t < 4; ++t)
#pragma unroll
    for (int i = 0; i < 8; ++i) acc[t][i] = 0.0f;
  for (int ksi = 0; ksi < kfrags; ++ksi) {
    h16 af = loadAfrag(aRow, ksi * 32 + hs);
#pragma unroll
    for (int t = 0; t < 4; ++t) {
      h16 bf = loadPackedB(pWf, t * kfrags + ksi, lane);
      acc[t] = wmma_f16(af, bf, acc[t]);
    }
  }
#pragma unroll
  for (int t = 0; t < 4; ++t) {
    unsigned n = (unsigned)(t * 16 + l16);
    float bev = be[n];
    float av = attn[n];  // attn[h][of] with h == t, of == l16
#pragma unroll
    for (int r = 0; r < 8; ++r) {
      unsigned e = e0 + (unsigned)(r + hs);
      unsigned si = (unsigned)src[e], di = (unsigned)dst[e];
      float f = acc[t][r] + aN[si * 64u + n] + bN[di * 64u + n] + bev;
      f = (f > 0.0f) ? f : f * 0.2f;  // leaky_relu, slope 0.2
      he_out[(size_t)(e * 64u + n)] = relu ? fmaxf(f, 0.0f) : f;
      float v = reduce16(f * av);
      if (l16 == 0) elog[(size_t)(e * 4u + (unsigned)t)] = v;  // lane0 -> r, lane16 -> r+8
    }
  }
}

// ---- edge softmax pieces ----
__global__ void seg_max_kernel(const float* __restrict__ elog, const int* __restrict__ dst,
                               float* __restrict__ m) {
  long i = (long)blockIdx.x * blockDim.x + threadIdx.x;
  if (i >= (long)NE * 4) return;
  unsigned e = (unsigned)(i >> 2); int h = (int)(i & 3);
  __hip_atomic_fetch_max(&m[(unsigned)dst[e] * 4u + (unsigned)h], elog[i],
                         __ATOMIC_RELAXED, __HIP_MEMORY_SCOPE_AGENT);
}

__global__ void exp_sum_kernel(float* __restrict__ elog, const int* __restrict__ dst,
                               const float* __restrict__ m, float* __restrict__ z) {
  long i = (long)blockIdx.x * blockDim.x + threadIdx.x;
  if (i >= (long)NE * 4) return;
  unsigned e = (unsigned)(i >> 2); int h = (int)(i & 3);
  float v = __expf(elog[i] - m[(unsigned)dst[e] * 4u + (unsigned)h]);
  elog[i] = v;
  __hip_atomic_fetch_add(&z[(unsigned)dst[e] * 4u + (unsigned)h], v,
                         __ATOMIC_RELAXED, __HIP_MEMORY_SCOPE_AGENT);
}

// ---- message scatter: h_out[dst] += hsrc[src] * (a/z[dst]) ----
__global__ void scatter_kernel(const float* __restrict__ hsrc, const float* __restrict__ a,
                               const float* __restrict__ z, const int* __restrict__ src,
                               const int* __restrict__ dst, float* __restrict__ hn_next) {
  long i = (long)blockIdx.x * blockDim.x + threadIdx.x;
  if (i >= (long)NE * 128) return;
  unsigned e = (unsigned)(i >> 7); unsigned c = (unsigned)(i & 127); unsigned h = c >> 5;
  unsigned di = (unsigned)dst[e];
  float w = a[e * 4u + h] / z[di * 4u + h];
  float v = hsrc[(unsigned)src[e] * 128u + c] * w;
  __hip_atomic_fetch_add(&hn_next[di * 128u + c], v,
                         __ATOMIC_RELAXED, __HIP_MEMORY_SCOPE_AGENT);
}

// ---- Fused link MLP: relu([hn[src]|hn[dst]|he] @ W1 + b1) @ W2 + b2 ----
// K = 320 in 10 steps of 32; chunk boundaries (128,256) are 8-aligned so the
// two 8-float A-chunks each live in exactly one piece.
__global__ void link_kernel(const float* __restrict__ hn, const float* __restrict__ he,
                            const int* __restrict__ src, const int* __restrict__ dst,
                            const _Float16* __restrict__ pW1, const float* __restrict__ b1,
                            const float* __restrict__ W2, const float* __restrict__ b2,
                            float* __restrict__ out) {
  int lane = threadIdx.x & 31;
  long wid = (long)(blockIdx.x * blockDim.x + threadIdx.x) >> 5;
  if (wid >= NE / 16) return;
  unsigned e0 = (unsigned)wid * 16u;
  int l16 = lane & 15;
  int hs = (lane & 16) ? 8 : 0;
  unsigned ea = e0 + (unsigned)l16;  // this lane's A-row edge
  const float* ps = hn + (size_t)((unsigned)src[ea] * 128u);
  const float* pd = hn + (size_t)((unsigned)dst[ea] * 128u);
  const float* pe = he + (size_t)(ea * 64u);
  f8 acc[4];
#pragma unroll
  for (int t = 0; t < 4; ++t)
#pragma unroll
    for (int i = 0; i < 8; ++i) acc[t][i] = 0.0f;
#pragma unroll
  for (int ksi = 0; ksi < 10; ++ksi) {
    int kb = ksi * 32 + hs;
    h16 af;
#pragma unroll
    for (int c = 0; c < 2; ++c) {
      int base = kb + c * 16;
      const float* p = (base < 128) ? (ps + base)
                     : (base < 256) ? (pd + base - 128)
                                    : (pe + base - 256);
#pragma unroll
      for (int j = 0; j < 8; ++j) af[c * 8 + j] = (_Float16)p[j];
    }
#pragma unroll
    for (int t = 0; t < 4; ++t) {
      h16 bf = loadPackedB(pW1, t * 10 + ksi, lane);
      acc[t] = wmma_f16(af, bf, acc[t]);
    }
  }
  float b2s = b2[0];
#pragma unroll
  for (int r = 0; r < 8; ++r) {
    float s = 0.0f;
#pragma unroll
    for (int t = 0; t < 4; ++t) {
      int n = t * 16 + l16;
      float v = fmaxf(acc[t][r] + b1[n], 0.0f);
      s += v * W2[n];
    }
    s = reduce16(s);
    if (l16 == 0) out[e0 + (unsigned)(r + hs)] = s + b2s;
  }
}

// ---- elementwise utilities ----
__global__ void fill_kernel(float* __restrict__ p, float v, long n) {
  long i = (long)blockIdx.x * blockDim.x + threadIdx.x;
  if (i < n) p[i] = v;
}
__global__ void relu_kernel(float* __restrict__ p, long n) {
  long i = (long)blockIdx.x * blockDim.x + threadIdx.x;
  if (i < n) p[i] = fmaxf(p[i], 0.0f);
}

extern "C" void kernel_launch(void* const* d_in, const int* in_sizes, int n_in,
                              void* d_out, int out_size, void* d_ws, size_t ws_size,
                              hipStream_t stream) {
  const float* x0 = (const float*)d_in[0];
  const float* e0 = (const float*)d_in[1];
  const int* src = (const int*)d_in[2];
  const int* dst = (const int*)d_in[3];
  auto W = [&](int i) { return (const float*)d_in[i]; };

  float* ws = (float*)d_ws;
  size_t off = 0;
  float* he   = ws + off; off += (size_t)NE * 64;
  float* hnA  = ws + off; off += (size_t)NN * 128;
  float* hnB  = ws + off; off += (size_t)NN * 128;
  float* aN   = ws + off; off += (size_t)NN * 64;
  float* bN   = ws + off; off += (size_t)NN * 64;
  float* cN   = ws + off; off += (size_t)NN * 128;
  float* elog = ws + off; off += (size_t)NE * 4;
  float* mbuf = ws + off; off += (size_t)NN * 4;
  float* zbuf = ws + off; off += (size_t)NN * 4;
  off = (off + 15) & ~(size_t)15;  // 64B-align packed-weight region

  // Packed f16 weight fragments (tiny; L2-resident).
  auto packedLen = [](int K, int N) { return (size_t)((N + 15) / 16) * (K / 32) * 512; };
  _Float16* hb = (_Float16*)(ws + off);
  size_t hoff = 0;
  _Float16 *pWni[4], *pWnj[4], *pWfij[4], *pWn[4];
  for (int l = 0; l < 4; ++l) {
    int Kn = l ? 128 : 64, Ke = l ? 64 : 32;
    pWni[l]  = hb + hoff; hoff += packedLen(Kn, 64);
    pWnj[l]  = hb + hoff; hoff += packedLen(Kn, 64);
    pWfij[l] = hb + hoff; hoff += packedLen(Ke, 64);
    pWn[l]   = hb + hoff; hoff += packedLen(Kn, 128);
  }
  _Float16* pNC = hb + hoff; hoff += packedLen(128, 16);
  _Float16* pEC = hb + hoff; hoff += packedLen(64, 8);
  _Float16* pW1 = hb + hoff; hoff += packedLen(320, 64);

  auto pack = [&](const float* Wm, _Float16* P, int K, int N) {
    long total = (long)packedLen(K, N);
    pack_b_kernel<<<dim3((unsigned)((total + 255) / 256)), 256, 0, stream>>>(Wm, P, K, N);
  };
  for (int l = 0; l < 4; ++l) {
    int base = 4 + l * 7;
    int Kn = l ? 128 : 64, Ke = l ? 64 : 32;
    pack(W(base + 0), pWni[l], Kn, 64);
    pack(W(base + 1), pWnj[l], Kn, 64);
    pack(W(base + 2), pWfij[l], Ke, 64);
    pack(W(base + 3), pWn[l], Kn, 128);
  }
  pack(W(32), pNC, 128, 16);
  pack(W(34), pEC, 64, 8);
  pack(W(36), pW1, 320, 64);

  auto gemm = [&](const float* A, const _Float16* Bp, const float* bias, float* C,
                  int M, int N, int K, int relu) {
    long waves = (long)(M / 16) * ((N + 15) / 16);
    dim3 grid((unsigned)((waves + 3) / 4));
    gemm_wmma_kernel<<<grid, 128, 0, stream>>>(A, Bp, bias, C, M, N, K, relu);
  };
  auto eblocks = [](long n) { return dim3((unsigned)((n + 255) / 256)); };
  dim3 egrid((unsigned)((NE / 16 + 3) / 4));  // wave-per-16-edges kernels

  float* hnCur = hnA;
  float* hnNext = hnB;

  for (int layer = 0; layer < 4; ++layer) {
    int base = 4 + layer * 7;
    const float* bn = W(base + 4);
    const float* attn = W(base + 5);
    const float* be = W(base + 6);
    const float* xin = layer ? hnCur : x0;
    int Kn = layer ? 128 : 64;
    const float* ein = layer ? he : e0;
    int Ke = layer ? 64 : 32;
    int relu = (layer < 3) ? 1 : 0;

    gemm(xin, pWni[layer], nullptr, aN, NN, 64, Kn, 0);
    gemm(xin, pWnj[layer], nullptr, bN, NN, 64, Kn, 0);
    gemm(xin, pWn[layer], bn, cN, NN, 128, Kn, 0);

    fill_kernel<<<eblocks((long)NN * 4), 256, 0, stream>>>(mbuf, -INFINITY, (long)NN * 4);
    fill_kernel<<<eblocks((long)NN * 4), 256, 0, stream>>>(zbuf, 0.0f, (long)NN * 4);
    fill_kernel<<<eblocks((long)NN * 128), 256, 0, stream>>>(hnNext, 0.0f, (long)NN * 128);

    egat_edge_kernel<<<egrid, 128, 0, stream>>>(ein, Ke, pWfij[layer], aN, bN, be, attn,
                                                src, dst, he, elog, relu);
    seg_max_kernel<<<eblocks((long)NE * 4), 256, 0, stream>>>(elog, dst, mbuf);
    exp_sum_kernel<<<eblocks((long)NE * 4), 256, 0, stream>>>(elog, dst, mbuf, zbuf);
    scatter_kernel<<<eblocks((long)NE * 128), 256, 0, stream>>>(cN, elog, zbuf, src, dst, hnNext);
    if (layer < 3)
      relu_kernel<<<eblocks((long)NN * 128), 256, 0, stream>>>(hnNext, (long)NN * 128);

    float* t = hnCur; hnCur = hnNext; hnNext = t;
  }

  float* out = (float*)d_out;
  float* out_node = out;                         // [50000, 16]
  float* out_edge = out + (size_t)NN * 16;       // [1.6M, 8]
  float* out_link = out_edge + (size_t)NE * 8;   // [1.6M, 1]

  gemm(hnCur, pNC, W(33), out_node, NN, 16, 128, 0);
  gemm(he, pEC, W(35), out_edge, NE, 8, 64, 0);
  link_kernel<<<egrid, 128, 0, stream>>>(hnCur, he, src, dst,
                                         pW1, W(37), W(38), W(39), out_link);
}
